// Indexer_33371895890037
// MI455X (gfx1250) — matrix operand order repaired and checked by reading, not compiled, not run on hardware
//
#include <hip/hip_runtime.h>
#include <hip/hip_bf16.h>
#include <stdint.h>

// ---------------------------------------------------------------------------
// CDNA5 / gfx1250 wave32 WMMA implementation of the DeepSeek-style indexer.
//   q = ql @ wq_b ; k = LN(hs @ wk) ; w = hs @ w_proj ; RoPE(q,k)
//   scores[t,s] = sum_h relu(q[t,h,:]·k[s,:]) * w[t,h] * 1/64  (causal)
//   top-256 per row (value desc, index asc tiebreak)
// All GEMMs: v_wmma_f32_16x16x32_bf16, f32 accumulation.
// Round 2: LDS-blocked q GEMM (128x128 tiles, double-buffered B) and
// LDS-staged per-head q tiles in the scores kernel to cut L2 traffic ~8x.
// ---------------------------------------------------------------------------

typedef __attribute__((ext_vector_type(16))) __bf16 v16bf;
typedef __attribute__((ext_vector_type(8)))  __bf16 v8bf;
typedef __attribute__((ext_vector_type(8)))  float  v8f;

#define T_DIM 2048
#define HID   4096
#define QLR   1536
#define NH    32
#define HD    128
#define TOPK  256

__device__ __forceinline__ unsigned short f2bf(float f) {
  unsigned int u = __float_as_uint(f);
  unsigned int r = u + 0x7FFFu + ((u >> 16) & 1u);   // round-to-nearest-even
  return (unsigned short)(r >> 16);
}

// A-fragment (16x32 bf16, M=lane%16): lanes 0-15 hold K 0..15,
// lanes 16-31 hold K 16..31. `chunk` points at row's K-chunk base.
__device__ __forceinline__ v16bf load_a_frag(const unsigned short* chunk, int lane) {
  const int off = (lane >= 16) ? 16 : 0;
  v8bf lo = *(const v8bf*)(chunk + off);
  v8bf hi = *(const v8bf*)(chunk + off + 8);
  v16bf a;
#pragma unroll
  for (int i = 0; i < 8; ++i) { a[i] = lo[i]; a[i + 8] = hi[i]; }
  return a;
}

// B-fragment (32x16 bf16, N=lane%16): lane holds contiguous K 0..15 (lanes<16)
// or K 16..31 (lanes>=16) of its column. `colchunk` = B^T row (contiguous K).
// Split into two 16B loads so only 16B alignment is required (LDS-friendly).
__device__ __forceinline__ v16bf load_b_frag(const unsigned short* colchunk, int lane) {
  const unsigned short* p = colchunk + ((lane >= 16) ? 16 : 0);
  v8bf lo = *(const v8bf*)(p);
  v8bf hi = *(const v8bf*)(p + 8);
  v16bf b;
#pragma unroll
  for (int i = 0; i < 8; ++i) { b[i] = lo[i]; b[i + 8] = hi[i]; }
  return b;
}

__device__ __forceinline__ v8f wmma_bf16(v16bf a, v16bf b, v8f c) {
  return __builtin_amdgcn_wmma_f32_16x16x32_bf16(false, a, false, b, (short)0, c,
                                                 false, false);
}

// ------------------------------ conversions --------------------------------
__global__ void cvt_kernel(const float* __restrict__ src,
                           unsigned short* __restrict__ dst, int n) {
  int i = blockIdx.x * blockDim.x + threadIdx.x;
  if (i < n) dst[i] = f2bf(src[i]);
}

// dst[c*rows + r] = bf16(src[r*cols + c])   (transpose while converting)
__global__ void cvtT_kernel(const float* __restrict__ src,
                            unsigned short* __restrict__ dst, int rows, int cols) {
  int i = blockIdx.x * blockDim.x + threadIdx.x;
  if (i < rows * cols) {
    int r = i / cols, c = i % cols;
    dst[c * rows + r] = f2bf(src[i]);
  }
}

// ------------------------- k = LN(hs@wk)+RoPE, w = hs@w_proj ---------------
// 10 waves: waves 0-7 -> 16x128 k tile, waves 8-9 -> 16x32 w tile.
__global__ __launch_bounds__(320) void k_w_kernel(
    const unsigned short* __restrict__ hsb, const unsigned short* __restrict__ wkT,
    const unsigned short* __restrict__ wpT, const float* __restrict__ gamma,
    const float* __restrict__ beta, const float* __restrict__ cosT,
    const float* __restrict__ sinT, unsigned short* __restrict__ kb,
    float* __restrict__ wf) {
  __shared__ float kbuf[16][HD];
  const int t0 = blockIdx.x * 16;
  const int tid = threadIdx.x, wv = tid >> 5, lane = tid & 31;
  const int row = t0 + (lane & 15);
  const unsigned short* arow = hsb + (size_t)row * HID;
  const int mhi = (lane >> 4) * 8;

  if (wv < 8) {
    const int n = wv * 16 + (lane & 15);
    const unsigned short* bcol = wkT + (size_t)n * HID;
    v8f c = {};
    for (int kc = 0; kc < HID / 32; ++kc)
      c = wmma_bf16(load_a_frag(arow + kc * 32, lane),
                    load_b_frag(bcol + kc * 32, lane), c);
#pragma unroll
    for (int r = 0; r < 8; ++r) kbuf[r + mhi][wv * 16 + (lane & 15)] = c[r];
  } else {
    const int n = (wv - 8) * 16 + (lane & 15);
    const unsigned short* bcol = wpT + (size_t)n * HID;
    v8f c = {};
    for (int kc = 0; kc < HID / 32; ++kc)
      c = wmma_bf16(load_a_frag(arow + kc * 32, lane),
                    load_b_frag(bcol + kc * 32, lane), c);
#pragma unroll
    for (int r = 0; r < 8; ++r) wf[(size_t)(t0 + r + mhi) * NH + n] = c[r];
  }
  __syncthreads();

  if (tid < 16) {                 // layernorm + RoPE, one row per thread
    const int t = t0 + tid;
    float mu = 0.f;
    for (int d = 0; d < HD; ++d) mu += kbuf[tid][d];
    mu *= (1.f / HD);
    float var = 0.f;
    for (int d = 0; d < HD; ++d) { float df = kbuf[tid][d] - mu; var += df * df; }
    var *= (1.f / HD);
    const float rs = rsqrtf(var + 1e-6f);
    for (int d = 0; d < HD; ++d)
      kbuf[tid][d] = (kbuf[tid][d] - mu) * rs * gamma[d] + beta[d];
    float out[64];
    for (int i = 0; i < 32; ++i) {          // interleaved RoPE on first 64 dims
      float e = kbuf[tid][2 * i], o = kbuf[tid][2 * i + 1];
      out[i]      = e * cosT[t * 64 + i]      - o * sinT[t * 64 + i];
      out[i + 32] = o * cosT[t * 64 + i + 32] + e * sinT[t * 64 + i + 32];
    }
    for (int i = 0; i < 64; ++i) kbuf[tid][i] = out[i];
    for (int d = 0; d < HD; ++d) kb[(size_t)t * HD + d] = f2bf(kbuf[tid][d]);
  }
}

// ----------------------- q GEMM: 128x128 LDS-blocked -----------------------
// Block computes rows t0..t0+127 x cols n0..n0+127 of q = ql @ wq_b.
// 8 waves; wave wv owns 16-row strip, all 8 n-tiles (acc: 8 x v8f).
// B k-chunks (128 cols x 32 k) double-buffered in LDS; A read from global
// (wave strips are disjoint -> no duplicated A traffic).
__global__ __launch_bounds__(256) void qgemm_kernel(
    const unsigned short* __restrict__ qlb, const unsigned short* __restrict__ wqbT,
    float* __restrict__ qraw) {
  __shared__ unsigned short bs[2][128][40];   // 40-stride: 16B-aligned, no conflicts
  const int tid = threadIdx.x, wv = tid >> 5, lane = tid & 31;
  const int t0 = blockIdx.x * 128, n0 = blockIdx.y * 128;
  const int arow_idx = t0 + wv * 16 + (lane & 15);
  const unsigned short* arow = qlb + (size_t)arow_idx * QLR;
  const int mhi = (lane >> 4) * 8;
  v8f acc[8] = {};

  const int sc_ = tid >> 1, sh_ = tid & 1;    // stage: 128 cols x 2 halves
  const unsigned short* ssrc = wqbT + (size_t)(n0 + sc_) * QLR + sh_ * 16;

  // stage chunk 0
  {
    v8bf x0 = *(const v8bf*)(ssrc);
    v8bf x1 = *(const v8bf*)(ssrc + 8);
    *(v8bf*)&bs[0][sc_][sh_ * 16]     = x0;
    *(v8bf*)&bs[0][sc_][sh_ * 16 + 8] = x1;
  }
  __syncthreads();

  for (int kc = 0; kc < QLR / 32; ++kc) {
    const int buf = kc & 1;
    if (kc + 1 < QLR / 32) {                  // stage next chunk into other buf
      const unsigned short* s = ssrc + (kc + 1) * 32;
      v8bf x0 = *(const v8bf*)(s);
      v8bf x1 = *(const v8bf*)(s + 8);
      *(v8bf*)&bs[buf ^ 1][sc_][sh_ * 16]     = x0;
      *(v8bf*)&bs[buf ^ 1][sc_][sh_ * 16 + 8] = x1;
    }
    __builtin_prefetch(arow + (kc + 2) * 32, 0, 0);
    v16bf a = load_a_frag(arow + kc * 32, lane);
#pragma unroll
    for (int nt = 0; nt < 8; ++nt) {
      v16bf b = load_b_frag(&bs[buf][nt * 16 + (lane & 15)][0], lane);
      acc[nt] = wmma_bf16(a, b, acc[nt]);
    }
    __syncthreads();
  }

#pragma unroll
  for (int nt = 0; nt < 8; ++nt)
#pragma unroll
    for (int r = 0; r < 8; ++r)
      qraw[(size_t)(t0 + wv * 16 + r + mhi) * HID + n0 + nt * 16 + (lane & 15)] =
          acc[nt][r];
}

// --------------------------- RoPE on q + bf16 ------------------------------
// grid (t, h), 128 threads: thread d produces output dim d of head h.
__global__ __launch_bounds__(128) void ropeq_kernel(
    const float* __restrict__ qraw, const float* __restrict__ cosT,
    const float* __restrict__ sinT, unsigned short* __restrict__ qb) {
  const int t = blockIdx.x, h = blockIdx.y, d = threadIdx.x;
  const float* row = qraw + (size_t)t * HID + h * HD;
  float v;
  if (d < 32) {
    v = row[2 * d] * cosT[t * 64 + d] - row[2 * d + 1] * sinT[t * 64 + d];
  } else if (d < 64) {
    const int i = d - 32;
    v = row[2 * i + 1] * cosT[t * 64 + d] + row[2 * i] * sinT[t * 64 + d];
  } else {
    v = row[d];
  }
  qb[(size_t)t * HID + h * HD + d] = f2bf(v);
}

// ------------------------------- scores ------------------------------------
// Block: (s_panel of 8 s_tiles, t_tile). 4 waves, wave wv owns 2 s_tiles
// (k fragments in registers). Per head, the 16x128 q tile is staged into LDS
// (double-buffered, 1 barrier/head) and shared by all waves.
__global__ __launch_bounds__(128) void scores_kernel(
    const unsigned short* __restrict__ qb, const unsigned short* __restrict__ kb,
    const float* __restrict__ wf, float* __restrict__ scores) {
  const int t_tile = blockIdx.y, panel = blockIdx.x;
  if (panel * 8 > t_tile) return;             // block-uniform early out
  __shared__ unsigned short qs[2][16][136];   // 136-stride: 16B-aligned rows
  __shared__ float wbuf[16][NH];
  const int tid = threadIdx.x, wv = tid >> 5, lane = tid & 31;
  const int t0 = t_tile * 16;

  for (int i = tid; i < 16 * NH; i += 128)
    wbuf[i >> 5][i & 31] = wf[(size_t)(t0 + (i >> 5)) * NH + (i & 31)];

  const int st0 = panel * 8 + wv * 2;         // wave's two s_tiles
  const bool act0 = (st0 <= t_tile);
  const bool act1 = (st0 + 1 <= t_tile);

  v16bf b[2][4];
#pragma unroll
  for (int u = 0; u < 2; ++u) {
    if (u == 0 ? act0 : act1) {
      const unsigned short* bbase = kb + (size_t)((st0 + u) * 16 + (lane & 15)) * HD;
#pragma unroll
      for (int ch = 0; ch < 4; ++ch) b[u][ch] = load_b_frag(bbase + ch * 32, lane);
    }
  }
  float acc[2][8] = {};

  // stage unit: 16 rows x 16 segments of 8 bf16; 128 threads -> 2 units each
  const int su0 = tid, su1 = tid + 128;
  const int sr0 = su0 >> 4, ss0 = su0 & 15;
  const int sr1 = su1 >> 4, ss1 = su1 & 15;

  // stage h = 0
  {
    *(v8bf*)&qs[0][sr0][ss0 * 8] =
        *(const v8bf*)(qb + (size_t)(t0 + sr0) * HID + ss0 * 8);
    *(v8bf*)&qs[0][sr1][ss1 * 8] =
        *(const v8bf*)(qb + (size_t)(t0 + sr1) * HID + ss1 * 8);
  }
  __syncthreads();

  for (int h = 0; h < NH; ++h) {
    const int cur = h & 1;
    if (h + 1 < NH) {                         // stage next head into other buf
      const size_t hb = (size_t)(h + 1) * HD;
      *(v8bf*)&qs[cur ^ 1][sr0][ss0 * 8] =
          *(const v8bf*)(qb + (size_t)(t0 + sr0) * HID + hb + ss0 * 8);
      *(v8bf*)&qs[cur ^ 1][sr1][ss1 * 8] =
          *(const v8bf*)(qb + (size_t)(t0 + sr1) * HID + hb + ss1 * 8);
    }
    v16bf a0 = load_a_frag(&qs[cur][lane & 15][0],  lane);
    v16bf a1 = load_a_frag(&qs[cur][lane & 15][32], lane);
    v16bf a2 = load_a_frag(&qs[cur][lane & 15][64], lane);
    v16bf a3 = load_a_frag(&qs[cur][lane & 15][96], lane);
#pragma unroll
    for (int u = 0; u < 2; ++u) {
      if (u == 0 ? act0 : act1) {             // wave-uniform
        v8f c = {};
        c = wmma_bf16(a0, b[u][0], c);
        c = wmma_bf16(a1, b[u][1], c);
        c = wmma_bf16(a2, b[u][2], c);
        c = wmma_bf16(a3, b[u][3], c);
        const int mhi = (lane >> 4) * 8;
#pragma unroll
        for (int r = 0; r < 8; ++r) {
          float v = c[r];
          v = v > 0.f ? v : 0.f;              // relu
          acc[u][r] += v * wbuf[r + mhi][h];  // * w[t,h]
        }
      }
    }
    __syncthreads();
  }

  const int mhi = (lane >> 4) * 8;
#pragma unroll
  for (int u = 0; u < 2; ++u) {
    if (u == 0 ? act0 : act1) {
      const int scol = (st0 + u) * 16 + (lane & 15);
#pragma unroll
      for (int r = 0; r < 8; ++r)
        scores[(size_t)(t0 + r + mhi) * T_DIM + scol] = acc[u][r] * 0.015625f;
    }
  }
}

// -------------------------------- top-k ------------------------------------
// One block per row. Bitonic sort on 64-bit keys: (value desc, index asc).
__global__ __launch_bounds__(256) void topk_kernel(
    const float* __restrict__ scores, const int* __restrict__ ks,
    const int* __restrict__ ke, int* __restrict__ idx_out,
    float* __restrict__ val_out) {
  __shared__ unsigned long long keys[T_DIM];
  const int t = blockIdx.x, tid = threadIdx.x;
  const int lo = ks[t], hi = ke[t];
  for (int j = tid; j < T_DIM; j += 256) {
    float v = (j >= lo && j < hi) ? scores[(size_t)t * T_DIM + j] : -1e30f;
    unsigned int u = __float_as_uint(v);
    unsigned int s = (u & 0x80000000u) ? ~u : (u | 0x80000000u); // ascending map
    unsigned int d = ~s;                                          // descending
    keys[j] = ((unsigned long long)d << 32) | (unsigned int)j;
  }
  __syncthreads();
  for (unsigned k = 2; k <= T_DIM; k <<= 1) {
    for (unsigned j = k >> 1; j > 0; j >>= 1) {
      for (int s = 0; s < T_DIM / 512; ++s) {
        unsigned i = tid + s * 256;
        unsigned ixj = i ^ j;
        if (ixj > i) {
          unsigned long long a = keys[i], b = keys[ixj];
          bool asc = ((i & k) == 0);
          if (asc ? (a > b) : (a < b)) { keys[i] = b; keys[ixj] = a; }
        }
      }
      __syncthreads();
    }
  }
  // first TOPK entries are the answer (sorted best-first)
  unsigned long long kk = keys[tid];
  unsigned int d = (unsigned int)(kk >> 32);
  unsigned int s = ~d;
  unsigned int u = (s & 0x80000000u) ? (s ^ 0x80000000u) : ~s;
  idx_out[(size_t)t * TOPK + tid] = (int)(kk & 0xFFFFFFFFu);
  val_out[(size_t)t * TOPK + tid] = __uint_as_float(u);
}

// ------------------------------- launcher ----------------------------------
extern "C" void kernel_launch(void* const* d_in, const int* in_sizes, int n_in,
                              void* d_out, int out_size, void* d_ws, size_t ws_size,
                              hipStream_t stream) {
  const float* hs    = (const float*)d_in[0];   // (1,2048,4096)
  const float* ql    = (const float*)d_in[1];   // (1,2048,1536)
  const int*   ks    = (const int*)  d_in[2];   // (2048,)
  const int*   ke    = (const int*)  d_in[3];   // (2048,)
  const float* cosT  = (const float*)d_in[4];   // (2048,64)
  const float* sinT  = (const float*)d_in[5];   // (2048,64)
  const float* wq_b  = (const float*)d_in[6];   // (1536,4096)
  const float* wk    = (const float*)d_in[7];   // (4096,128)
  const float* gamma = (const float*)d_in[8];   // (128,)
  const float* beta  = (const float*)d_in[9];   // (128,)
  const float* wproj = (const float*)d_in[10];  // (4096,32)

  char* ws = (char*)d_ws;
  size_t off = 0;
  auto alloc = [&](size_t bytes) { char* p = ws + off; off += (bytes + 255) & ~(size_t)255; return p; };
  unsigned short* hsb  = (unsigned short*)alloc((size_t)T_DIM * HID * 2);
  unsigned short* qlb  = (unsigned short*)alloc((size_t)T_DIM * QLR * 2);
  unsigned short* wqbT = (unsigned short*)alloc((size_t)HID * QLR * 2);    // (4096,1536)
  unsigned short* wkT  = (unsigned short*)alloc((size_t)HD * HID * 2);     // (128,4096)
  unsigned short* wpT  = (unsigned short*)alloc((size_t)NH * HID * 2);     // (32,4096)
  unsigned short* qb   = (unsigned short*)alloc((size_t)T_DIM * NH * HD * 2);
  unsigned short* kb   = (unsigned short*)alloc((size_t)T_DIM * HD * 2);
  float*          wf   = (float*)alloc((size_t)T_DIM * NH * 4);
  float*          qraw = (float*)alloc((size_t)T_DIM * HID * 4);
  float*          sc   = (float*)alloc((size_t)T_DIM * T_DIM * 4);

  const int CB = 256;
  cvt_kernel <<<(T_DIM * HID + CB - 1) / CB, CB, 0, stream>>>(hs, hsb, T_DIM * HID);
  cvt_kernel <<<(T_DIM * QLR + CB - 1) / CB, CB, 0, stream>>>(ql, qlb, T_DIM * QLR);
  cvtT_kernel<<<(QLR * HID + CB - 1) / CB, CB, 0, stream>>>(wq_b, wqbT, QLR, HID);
  cvtT_kernel<<<(HID * HD  + CB - 1) / CB, CB, 0, stream>>>(wk, wkT, HID, HD);
  cvtT_kernel<<<(HID * NH  + CB - 1) / CB, CB, 0, stream>>>(wproj, wpT, HID, NH);

  k_w_kernel<<<T_DIM / 16, 320, 0, stream>>>(hsb, wkT, wpT, gamma, beta, cosT, sinT,
                                             kb, wf);
  qgemm_kernel<<<dim3(T_DIM / 128, HID / 128), 256, 0, stream>>>(qlb, wqbT, qraw);
  ropeq_kernel<<<dim3(T_DIM, NH), 128, 0, stream>>>(qraw, cosT, sinT, qb);
  scores_kernel<<<dim3(T_DIM / 128, T_DIM / 16), 128, 0, stream>>>(qb, kb, wf, sc);

  int*   idx_out = (int*)d_out;                          // (1,2048,1,256) int32
  float* val_out = (float*)d_out + (size_t)T_DIM * TOPK; // (2048,256) f32
  topk_kernel<<<T_DIM, 256, 0, stream>>>(sc, ks, ke, idx_out, val_out);
}
